// MultiGranularityConsistencyNet_57921928954157
// MI455X (gfx1250) — matrix operand order repaired and checked by reading, not compiled, not run on hardware
//
#include <hip/hip_runtime.h>
#include <math.h>

#define BB 32
#define TT 64
#define PPN 196
#define DD 768
#define HH 12
#define HD 64
#define PPAD 224
#define PTILES 13

typedef __attribute__((ext_vector_type(16))) __bf16 v16bf;
typedef __attribute__((ext_vector_type(8)))  float  v8f;
typedef __attribute__((ext_vector_type(4)))  int    v4i;

union FragU { v16bf v; unsigned int u[8]; };

#if defined(__HIP_DEVICE_COMPILE__) && __has_builtin(__builtin_amdgcn_global_load_async_to_lds_b128)
#define ASYNC_LDS 1
#define AS1 __attribute__((address_space(1)))
#define AS3 __attribute__((address_space(3)))
#endif

__device__ __forceinline__ unsigned short f2bf(float x){
  unsigned int u = __float_as_uint(x);
  return (unsigned short)((u + 0x7fffu + ((u >> 16) & 1u)) >> 16);
}
__device__ __forceinline__ float bf2f(unsigned short s){
  return __uint_as_float(((unsigned int)s) << 16);
}
__device__ __forceinline__ v8f v8f_zero(){
  v8f z;
#pragma unroll
  for (int i = 0; i < 8; ++i) z[i] = 0.f;
  return z;
}
__device__ __forceinline__ v8f wmma_bf16(v16bf a, v16bf b, v8f c){
  return __builtin_amdgcn_wmma_f32_16x16x32_bf16(false, a, false, b, (short)0, c, false, false);
}

// ---------------- fp32 -> bf16 conversion (row-major copy) ----------------
__global__ __launch_bounds__(256) void k_cvt(const float* __restrict__ src,
                                             unsigned short* __restrict__ dst, int n){
  int i = blockIdx.x * 256 + threadIdx.x;
  if (i < n) dst[i] = f2bf(src[i]);
}

// ---------------- fp32 [K][N] -> bf16 transposed [N][K] (LDS-tiled, coalesced) ----------------
__global__ __launch_bounds__(256) void k_cvt_t(const float* __restrict__ W,
                                               unsigned short* __restrict__ WT, int K, int N){
  __shared__ float t[32][33];
  int bn = blockIdx.x * 32;     // N tile
  int bk = blockIdx.y * 32;     // K tile
  int tx = threadIdx.x & 31, ty = threadIdx.x >> 5;   // 32 x 8
#pragma unroll
  for (int r = 0; r < 32; r += 8)
    t[ty + r][tx] = W[(size_t)(bk + ty + r) * N + bn + tx];
  __syncthreads();
#pragma unroll
  for (int r = 0; r < 32; r += 8)
    WT[(size_t)(bn + ty + r) * K + bk + tx] = f2bf(t[tx][ty + r]);
}

// ---------------- generic bf16 WMMA GEMM: C = A[MxK] * W[KxN] (+bias) ----------------
// A row-major bf16 (lda=K); WT = W transposed, row-major [N][K] bf16.
// K % 32 == 0, N % 128 == 0. Double-buffered LDS, async global->LDS when available.
__global__ __launch_bounds__(256) void k_gemm(const unsigned short* __restrict__ A,
    const unsigned short* __restrict__ WT, const float* __restrict__ bias,
    float* __restrict__ Cf, unsigned short* __restrict__ Cb, int M, int N, int K)
{
  __shared__ unsigned short As0[128 * 32];   // [m][k]
  __shared__ unsigned short As1[128 * 32];
  __shared__ unsigned short Bs0[128 * 32];   // [n][k]
  __shared__ unsigned short Bs1[128 * 32];
  int tid = threadIdx.x, wave = tid >> 5, lane = tid & 31, half = lane >> 4, l16 = lane & 15;
  int bm = blockIdx.y * 128, bn = blockIdx.x * 128;
  int wm = (wave >> 1) * 32, wn = (wave & 1) * 64;

  // stage one 128x32 tile (per-thread share = 2 x 16B) into LDS
  auto issueA = [&](int k0, unsigned short* dst){
#pragma unroll
    for (int it = 0; it < 2; ++it){
      int idx = tid + it * 256;                 // 0..511
      int r = idx >> 2, c4 = (idx & 3) * 8;
      int gr = bm + r;
      const unsigned short* g = A + (size_t)gr * K + k0 + c4;
      unsigned short* l = dst + r * 32 + c4;
#ifdef ASYNC_LDS
      if (gr < M)
        __builtin_amdgcn_global_load_async_to_lds_b128((AS1 v4i*)g, (AS3 v4i*)l, 0, 0);
#else
      uint4 val;
      if (gr < M) val = *(const uint4*)g;
      else { val.x = 0u; val.y = 0u; val.z = 0u; val.w = 0u; }
      *(uint4*)l = val;
#endif
    }
  };
  auto issueB = [&](int k0, unsigned short* dst){
#pragma unroll
    for (int it = 0; it < 2; ++it){
      int idx = tid + it * 256;
      int r = idx >> 2, c4 = (idx & 3) * 8;
      const unsigned short* g = WT + (size_t)(bn + r) * K + k0 + c4;
      unsigned short* l = dst + r * 32 + c4;
#ifdef ASYNC_LDS
      __builtin_amdgcn_global_load_async_to_lds_b128((AS1 v4i*)g, (AS3 v4i*)l, 0, 0);
#else
      *(uint4*)l = *(const uint4*)g;
#endif
    }
  };

  v8f acc[2][4];
#pragma unroll
  for (int sm = 0; sm < 2; ++sm)
#pragma unroll
    for (int sn = 0; sn < 4; ++sn) acc[sm][sn] = v8f_zero();

  int nt = K >> 5;
  issueA(0, As0); issueB(0, Bs0);
  if (nt > 1){ issueA(32, As1); issueB(32, Bs1); }

  for (int i = 0; i < nt; ++i){
    unsigned short* as = (i & 1) ? As1 : As0;
    unsigned short* bs = (i & 1) ? Bs1 : Bs0;
#ifdef ASYNC_LDS
    // my wave issued 4 async ops per tile; <=4 left in flight => tile i complete
    if (i + 1 < nt) asm volatile("s_wait_asynccnt 0x4" ::: "memory");
    else            asm volatile("s_wait_asynccnt 0x0" ::: "memory");
#endif
    __syncthreads();

    FragU af[2], bfr[4];
#pragma unroll
    for (int sm = 0; sm < 2; ++sm){
      const unsigned short* ap = as + (wm + sm * 16 + l16) * 32 + half * 16;
#pragma unroll
      for (int j = 0; j < 8; ++j)
        af[sm].u[j] = *(const unsigned int*)(ap + 2 * j);
    }
#pragma unroll
    for (int sn = 0; sn < 4; ++sn){
      const unsigned short* bp = bs + (wn + sn * 16 + l16) * 32 + half * 16;
#pragma unroll
      for (int j = 0; j < 8; ++j)
        bfr[sn].u[j] = *(const unsigned int*)(bp + 2 * j);
    }
#pragma unroll
    for (int sm = 0; sm < 2; ++sm)
#pragma unroll
      for (int sn = 0; sn < 4; ++sn)
        acc[sm][sn] = wmma_bf16(af[sm].v, bfr[sn].v, acc[sm][sn]);
    __syncthreads();

    if (i + 2 < nt){ int k0 = (i + 2) * 32; issueA(k0, as); issueB(k0, bs); }
  }

#pragma unroll
  for (int sm = 0; sm < 2; ++sm)
#pragma unroll
    for (int sn = 0; sn < 4; ++sn){
      int gn = bn + wn + sn * 16 + l16;
      float bv = bias ? bias[gn] : 0.f;
#pragma unroll
      for (int r = 0; r < 8; ++r){
        int gm = bm + wm + sm * 16 + r + half * 8;
        if (gm < M){
          float v = acc[sm][sn][r] + bv;
          if (Cf) Cf[(size_t)gm * N + gn] = v;
          if (Cb) Cb[(size_t)gm * N + gn] = f2bf(v);
        }
      }
    }
}

// ---------------- cross attention: one WG per (b,h), 4 waves x 16 query rows ----------------
__global__ __launch_bounds__(128) void k_attn(const unsigned short* __restrict__ Qb,
    const unsigned short* __restrict__ Kb, const unsigned short* __restrict__ Vb,
    unsigned short* __restrict__ updb)
{
  __shared__ unsigned short attL[4 * 16 * PPAD];
  __shared__ unsigned short Vt[64 * PPAD];         // [d][p] transposed
  int b = blockIdx.x / HH, h = blockIdx.x % HH;
  int tid = threadIdx.x, wave = tid >> 5, lane = tid & 31, half = lane >> 4, l16 = lane & 15;
  size_t qrow0 = (size_t)b * TT, krow0 = (size_t)b * PPN;
  int dbase = h * HD;

  // stage V^T for this (b,h)
  for (int p = tid; p < PPAD; p += 128){
    if (p < PPN){
      const uint4* vr = (const uint4*)(Vb + (krow0 + p) * DD + dbase);
#pragma unroll
      for (int q = 0; q < 8; ++q){
        uint4 x = vr[q];
        const unsigned short* xs = (const unsigned short*)&x;
#pragma unroll
        for (int e = 0; e < 8; ++e) Vt[(q * 8 + e) * PPAD + p] = xs[e];
      }
    } else {
      for (int d = 0; d < 64; ++d) Vt[d * PPAD + p] = 0;
    }
  }

  // Q A-fragments (two K=32 chunks over HD=64), loaded straight from global
  FragU qa[2];
  {
    const unsigned short* qp = Qb + (qrow0 + wave * 16 + l16) * DD + dbase + half * 16;
#pragma unroll
    for (int c = 0; c < 2; ++c)
#pragma unroll
      for (int j = 0; j < 8; ++j)
        qa[c].u[j] = *(const unsigned int*)(qp + c * 32 + 2 * j);
  }

  // scores = Q K^T / 8 over 13 column tiles (P padded 196->208)
  v8f sc[PTILES];
#pragma unroll
  for (int pj = 0; pj < PTILES; ++pj){
    int p = pj * 16 + l16;
    bool ok = p < PPN;
    const unsigned short* kp = Kb + (krow0 + (ok ? p : 0)) * DD + dbase + half * 16;
    FragU b0, b1;
#pragma unroll
    for (int j = 0; j < 8; ++j){
      b0.u[j] = ok ? *(const unsigned int*)(kp + 2 * j) : 0u;
      b1.u[j] = ok ? *(const unsigned int*)(kp + 32 + 2 * j) : 0u;
    }
    v8f a = v8f_zero();
    a = wmma_bf16(qa[0].v, b0.v, a);
    a = wmma_bf16(qa[1].v, b1.v, a);
    sc[pj] = a;
  }

  // row softmax (each row spread over 16 lanes x 13 tiles) -> att bf16 into LDS
  unsigned short* abase = attL + wave * 16 * PPAD;
#pragma unroll
  for (int r = 0; r < 8; ++r){
    float m = -1e30f;
#pragma unroll
    for (int pj = 0; pj < PTILES; ++pj){
      float s = sc[pj][r] * 0.125f;
      if (pj * 16 + l16 >= PPN) s = -1e30f;
      sc[pj][r] = s;
      m = fmaxf(m, s);
    }
#pragma unroll
    for (int o = 1; o < 16; o <<= 1) m = fmaxf(m, __shfl_xor(m, o));
    float sum = 0.f;
#pragma unroll
    for (int pj = 0; pj < PTILES; ++pj){
      float e = (pj * 16 + l16 < PPN) ? __expf(sc[pj][r] - m) : 0.f;
      sc[pj][r] = e;
      sum += e;
    }
#pragma unroll
    for (int o = 1; o < 16; o <<= 1) sum += __shfl_xor(sum, o);
    float inv = 1.f / sum;
    unsigned short* arow = abase + (r + half * 8) * PPAD;
#pragma unroll
    for (int pj = 0; pj < PTILES; ++pj) arow[pj * 16 + l16] = f2bf(sc[pj][r] * inv);
  }
  // zero-pad columns 208..223 of this wave's 16 rows
  for (int q = lane; q < 256; q += 32){
    int rr = q >> 4, cc = q & 15;
    abase[rr * PPAD + 208 + cc] = 0;
  }
  __syncthreads();

  // upd = att (16x224) x V (224x64)
  v8f up[4];
#pragma unroll
  for (int sn = 0; sn < 4; ++sn) up[sn] = v8f_zero();
  for (int kc = 0; kc < 7; ++kc){
    FragU af;
    const unsigned short* ap = abase + l16 * PPAD + kc * 32 + half * 16;
#pragma unroll
    for (int j = 0; j < 8; ++j)
      af.u[j] = *(const unsigned int*)(ap + 2 * j);
#pragma unroll
    for (int sn = 0; sn < 4; ++sn){
      FragU bv;
      const unsigned short* vp = Vt + (sn * 16 + l16) * PPAD + kc * 32 + half * 16;
#pragma unroll
      for (int j = 0; j < 8; ++j) bv.u[j] = *(const unsigned int*)(vp + 2 * j);
      up[sn] = wmma_bf16(af.v, bv.v, up[sn]);
    }
  }
#pragma unroll
  for (int sn = 0; sn < 4; ++sn)
#pragma unroll
    for (int r = 0; r < 8; ++r){
      int t = wave * 16 + r + half * 8;
      updb[(qrow0 + t) * DD + dbase + sn * 16 + l16] = f2bf(up[sn][r]);
    }
}

// ---------------- per-head source/dest GAT scores: src=Wh.a1, dst=Wh.a2 ----------------
__global__ __launch_bounds__(256) void k_srcdst(const unsigned short* __restrict__ Whb,
    const float* __restrict__ avec, float* __restrict__ srcv, float* __restrict__ dstv, int total)
{
  int idx = blockIdx.x * 256 + threadIdx.x;
  if (idx >= total * HH) return;
  int node = idx / HH, h = idx % HH;
  const unsigned short* wr = Whb + (size_t)node * DD + h * HD;
  float s = 0.f, d = 0.f;
#pragma unroll
  for (int f = 0; f < HD; ++f){
    float w = bf2f(wr[f]);
    s += w * avec[f];
    d += w * avec[HD + f];
  }
  srcv[idx] = s; dstv[idx] = d;
}

// ---------------- GAT: softmax(adj-masked leakyrelu) x Wh, ELU, atomic mean-accum ----------------
__global__ __launch_bounds__(128) void k_gat(const unsigned short* __restrict__ Whb,
    const float* __restrict__ srcv, const float* __restrict__ dstv,
    const int* __restrict__ adj, int adjStride,
    float* __restrict__ outAcc, int N, int NPAD, int KCH)
{
  __shared__ unsigned short attL[4 * 16 * PPAD];
  __shared__ unsigned short WhT[64 * PPAD];       // [f][j]
  __shared__ float invL[4 * 16];
  int b = blockIdx.x / HH, h = blockIdx.x % HH;
  int tid = threadIdx.x, wave = tid >> 5, lane = tid & 31, half = lane >> 4, l16 = lane & 15;
  size_t nb = (size_t)b * N;
  int dbase = h * HD;

  for (int n = tid; n < NPAD; n += 128){
    if (n < N){
      const uint4* wr = (const uint4*)(Whb + (nb + n) * DD + dbase);
#pragma unroll
      for (int q = 0; q < 8; ++q){
        uint4 x = wr[q];
        const unsigned short* xs = (const unsigned short*)&x;
#pragma unroll
        for (int e = 0; e < 8; ++e) WhT[(q * 8 + e) * NPAD + n] = xs[e];
      }
    } else {
      for (int d = 0; d < 64; ++d) WhT[d * NPAD + n] = 0;
    }
  }
  __syncthreads();

  unsigned short* abase = attL + wave * 16 * NPAD;
  int ntiles = (N + 15) >> 4;
  for (int tile = wave; tile < ntiles; tile += 4){
    int i = tile * 16 + l16;                // lanes l and l+16 share row i, split j by parity
    bool rv = i < N;
    const int* arow = adj + (size_t)b * adjStride + (size_t)(rv ? i : 0) * N;
    float si = rv ? srcv[(nb + i) * HH + h] : 0.f;
    float m = -1e30f;
    if (rv){
      for (int j = half; j < N; j += 2){
        if (arow[j] != 0){
          float e = si + dstv[(nb + j) * HH + h];
          e = (e >= 0.f) ? e : 0.2f * e;
          m = fmaxf(m, e);
        }
      }
    }
    m = fmaxf(m, __shfl_xor(m, 16));
    float sum = 0.f;
    unsigned short* arw = abase + l16 * NPAD;
    if (rv){
      for (int j = half; j < N; j += 2){
        float a = 0.f;
        if (arow[j] != 0){
          float e = si + dstv[(nb + j) * HH + h];
          e = (e >= 0.f) ? e : 0.2f * e;
          a = __expf(e - m);
        }
        sum += a;
        arw[j] = f2bf(a);               // unnormalized; 1/sum applied after WMMA
      }
      for (int j = N + half; j < NPAD; j += 2) arw[j] = 0;
    }
    sum += __shfl_xor(sum, 16);
    if (half == 0) invL[wave * 16 + l16] = rv ? (1.f / sum) : 0.f;
    asm volatile("s_wait_dscnt 0" ::: "memory");

    v8f hp[4];
#pragma unroll
    for (int sn = 0; sn < 4; ++sn) hp[sn] = v8f_zero();
    for (int kc = 0; kc < KCH; ++kc){
      FragU af;
      const unsigned short* ap = abase + l16 * NPAD + kc * 32 + half * 16;
#pragma unroll
      for (int j = 0; j < 8; ++j)
        af.u[j] = *(const unsigned int*)(ap + 2 * j);
#pragma unroll
      for (int sn = 0; sn < 4; ++sn){
        FragU bv;
        const unsigned short* bp = WhT + (sn * 16 + l16) * NPAD + kc * 32 + half * 16;
#pragma unroll
        for (int j = 0; j < 8; ++j) bv.u[j] = *(const unsigned int*)(bp + 2 * j);
        hp[sn] = wmma_bf16(af.v, bv.v, hp[sn]);
      }
    }
#pragma unroll
    for (int sn = 0; sn < 4; ++sn)
#pragma unroll
      for (int r = 0; r < 8; ++r){
        int irow = tile * 16 + r + half * 8;
        if (irow < N){
          float v = hp[sn][r] * invL[wave * 16 + r + half * 8];
          v = (v > 0.f) ? v : (__expf(v) - 1.f);
          atomicAdd(outAcc + b * DD + dbase + sn * 16 + l16, v);
        }
      }
    asm volatile("s_wait_dscnt 0" ::: "memory");
  }
}

// ---------------- fusion stage 1: s_token, f_token, cosines, scores3 ----------------
__device__ float blkred(float v, float* red){
  int tid = threadIdx.x;
  red[tid] = v; __syncthreads();
  for (int s = 128; s > 0; s >>= 1){ if (tid < s) red[tid] += red[tid + s]; __syncthreads(); }
  float r = red[0]; __syncthreads(); return r;
}

__global__ __launch_bounds__(256) void k_r1(const float* __restrict__ O,
    const float* __restrict__ img, float* __restrict__ tgfa, const float* __restrict__ igfa,
    const float* __restrict__ ct, const float* __restrict__ ci, const float* __restrict__ ls,
    float* __restrict__ ftok, float* __restrict__ sc3)
{
  __shared__ float red[256];
  int b = blockIdx.x, tid = threadIdx.x;
  float part = 0.f;
  for (int d = tid; d < DD; d += 256){
    float ft = 0.f, st = 0.f;
    for (int t = 0; t < TT; ++t){
      float o = O[((size_t)b * TT + t) * DD + d];
      ft += o;
      st += o * img[((size_t)b * PPN + t) * DD + d];
    }
    ftok[b * DD + d] = ft * (1.f / TT);
    part += st;
  }
  float stok = blkred(part, red) * (1.f / TT);
  float dp = 0.f, nt = 0.f, ni = 0.f;
  for (int d = tid; d < DD; d += 256){
    float tg = tgfa[b * DD + d] * (1.f / TT);
    float ig = igfa[b * DD + d] * (1.f / PPN);
    tgfa[b * DD + d] = tg;
    dp += tg * ig; nt += tg * tg; ni += ig * ig;
  }
  dp = blkred(dp, red); nt = blkred(nt, red); ni = blkred(ni, red);
  float sph = dp / (fmaxf(sqrtf(nt), 1e-8f) * fmaxf(sqrtf(ni), 1e-8f));
  float cdp = 0.f, cn1 = 0.f, cn2 = 0.f;
  for (int i2 = tid; i2 < 512; i2 += 256){
    float a = ct[b * 512 + i2], c = ci[b * 512 + i2];
    cdp += a * c; cn1 += a * a; cn2 += c * c;
  }
  cdp = blkred(cdp, red); cn1 = blkred(cn1, red); cn2 = blkred(cn2, red);
  float sg = expf(ls[0]) * cdp / (fmaxf(sqrtf(cn1), 1e-8f) * fmaxf(sqrtf(cn2), 1e-8f));
  if (tid == 0){ sc3[b * 3 + 0] = stok; sc3[b * 3 + 1] = sph; sc3[b * 3 + 2] = sg; }
}

// ---------------- fusion stage 2: gating MLP + f_agg (bf16 for c1 GEMM) ----------------
__global__ __launch_bounds__(256) void k_r2(const float* __restrict__ sc3,
    const float* __restrict__ f1w, const float* __restrict__ f1b,
    const float* __restrict__ f2w, const float* __restrict__ f2b,
    const float* __restrict__ ftok, const float* __restrict__ tgf,
    const float* __restrict__ fg, unsigned short* __restrict__ faggb)
{
  int b = blockIdx.x, tid = threadIdx.x;
  float s0 = sc3[b * 3 + 0], s1 = sc3[b * 3 + 1], s2 = sc3[b * 3 + 2];
  float g[16];
#pragma unroll
  for (int j = 0; j < 16; ++j){
    float z = s0 * f1w[j] + s1 * f1w[16 + j] + s2 * f1w[32 + j] + f1b[j];
    g[j] = 0.5f * z * (1.f + erff(z * 0.70710678118f));   // exact GELU
  }
  float w_[3];
#pragma unroll
  for (int o = 0; o < 3; ++o){
    float z = f2b[o];
#pragma unroll
    for (int j = 0; j < 16; ++j) z += g[j] * f2w[j * 3 + o];
    w_[o] = 1.f / (1.f + __expf(-z));
  }
  for (int d = tid; d < DD; d += 256){
    float fa = w_[0] * ftok[b * DD + d] + w_[1] * tgf[b * DD + d] + w_[2] * fg[b * DD + d];
    faggb[b * DD + d] = f2bf(fa);
  }
}

// ---------------- final head: relu(h1) @ c2w + c2b ----------------
__global__ __launch_bounds__(64) void k_r3(const float* __restrict__ h1,
    const float* __restrict__ c2w, const float* __restrict__ c2b, float* __restrict__ out)
{
  int idx = threadIdx.x;
  if (idx >= BB * 2) return;
  int b = idx >> 1, o = idx & 1;
  float acc = c2b[o];
  for (int k = 0; k < 384; ++k){
    float hv = fmaxf(h1[b * 384 + k], 0.f);
    acc += hv * c2w[k * 2 + o];
  }
  out[idx] = acc;
}

extern "C" void kernel_launch(void* const* d_in, const int* in_sizes, int n_in,
                              void* d_out, int out_size, void* d_ws, size_t ws_size,
                              hipStream_t stream)
{
  const float* text = (const float*)d_in[0];
  const float* img  = (const float*)d_in[1];
  const float* ct   = (const float*)d_in[2];
  const float* ci   = (const float*)d_in[3];
  const float* ls   = (const float*)d_in[4];
  const float* wq = (const float*)d_in[5];  const float* bq = (const float*)d_in[6];
  const float* wk = (const float*)d_in[7];  const float* bk = (const float*)d_in[8];
  const float* wv = (const float*)d_in[9];  const float* bv = (const float*)d_in[10];
  const float* wo = (const float*)d_in[11]; const float* bo = (const float*)d_in[12];
  const float* Wt = (const float*)d_in[13]; const float* at = (const float*)d_in[14];
  const float* Wi = (const float*)d_in[15]; const float* ai = (const float*)d_in[16];
  const float* Wp = (const float*)d_in[17]; const float* bp = (const float*)d_in[18];
  const float* f1w = (const float*)d_in[19]; const float* f1b = (const float*)d_in[20];
  const float* f2w = (const float*)d_in[21]; const float* f2b = (const float*)d_in[22];
  const float* c1w = (const float*)d_in[23]; const float* c1b = (const float*)d_in[24];
  const float* c2w = (const float*)d_in[25]; const float* c2b = (const float*)d_in[26];
  const int* tadj = (const int*)d_in[27];
  const int* iadj = (const int*)d_in[28];

  const size_t MT = 2048, MI = 6272;
  char* base = (char*)d_ws; size_t off = 0;
  auto alloc = [&](size_t bytes)->void*{
    void* p = base + off; off = (off + bytes + 255) & ~(size_t)255; return p;
  };
  unsigned short* textb = (unsigned short*)alloc(MT * DD * 2);
  unsigned short* imgb  = (unsigned short*)alloc(MI * DD * 2);
  unsigned short* wqb = (unsigned short*)alloc(768ull * 768 * 2);   // transposed [N][K]
  unsigned short* wkb = (unsigned short*)alloc(768ull * 768 * 2);
  unsigned short* wvb = (unsigned short*)alloc(768ull * 768 * 2);
  unsigned short* wob = (unsigned short*)alloc(768ull * 768 * 2);
  unsigned short* Wtb = (unsigned short*)alloc(768ull * 768 * 2);
  unsigned short* Wib = (unsigned short*)alloc(768ull * 768 * 2);
  unsigned short* Wpb = (unsigned short*)alloc(768ull * 512 * 2);   // [768][512]
  unsigned short* ctb = (unsigned short*)alloc(32ull * 512 * 2);
  unsigned short* c1wb = (unsigned short*)alloc(384ull * 768 * 2);  // [384][768]
  unsigned short* Qb = (unsigned short*)alloc(MT * DD * 2);
  unsigned short* Kb = (unsigned short*)alloc(MI * DD * 2);
  unsigned short* Vb = (unsigned short*)alloc(MI * DD * 2);
  unsigned short* WhTb = (unsigned short*)alloc(MT * DD * 2);
  unsigned short* WhIb = (unsigned short*)alloc(MI * DD * 2);
  unsigned short* updb = (unsigned short*)alloc(MT * DD * 2);
  float* Of   = (float*)alloc(MT * DD * 4);
  float* srcT = (float*)alloc(MT * HH * 4);
  float* dstT = (float*)alloc(MT * HH * 4);
  float* srcI = (float*)alloc(MI * HH * 4);
  float* dstI = (float*)alloc(MI * HH * 4);
  float* tgfa = (float*)alloc(32ull * DD * 4);
  float* igfa = (float*)alloc(32ull * DD * 4);
  float* fg   = (float*)alloc(32ull * DD * 4);
  float* ftok = (float*)alloc(32ull * DD * 4);
  float* sc3  = (float*)alloc(96 * 4);
  unsigned short* faggb = (unsigned short*)alloc(32ull * DD * 2);
  float* h1 = (float*)alloc(32ull * 384 * 4);

  auto cvt = [&](const float* s, unsigned short* d, int n){
    k_cvt<<<(n + 255) / 256, 256, 0, stream>>>(s, d, n);
  };
  auto cvtT = [&](const float* W, unsigned short* WTd, int K, int N){
    dim3 g(N / 32, K / 32);
    k_cvt_t<<<g, 256, 0, stream>>>(W, WTd, K, N);
  };
  auto gemm = [&](const unsigned short* Aa, const unsigned short* WTd, const float* bias,
                  float* Cf, unsigned short* Cb, int M, int N, int K){
    dim3 g(N / 128, (M + 127) / 128);
    k_gemm<<<g, 256, 0, stream>>>(Aa, WTd, bias, Cf, Cb, M, N, K);
  };

  cvt(text, textb, (int)(MT * DD));
  cvt(img,  imgb,  (int)(MI * DD));
  cvtT(wq, wqb, 768, 768);  cvtT(wk, wkb, 768, 768);
  cvtT(wv, wvb, 768, 768);  cvtT(wo, wob, 768, 768);
  cvtT(Wt, Wtb, 768, 768);  cvtT(Wi, Wib, 768, 768);
  cvtT(Wp, Wpb, 512, 768);  cvt(ct, ctb, 32 * 512);
  cvtT(c1w, c1wb, 768, 384);
  (void)hipMemsetAsync(tgfa, 0, 32ull * DD * 4, stream);
  (void)hipMemsetAsync(igfa, 0, 32ull * DD * 4, stream);

  gemm(textb, wqb, bq, nullptr, Qb, (int)MT, 768, 768);
  gemm(imgb,  wkb, bk, nullptr, Kb, (int)MI, 768, 768);
  gemm(imgb,  wvb, bv, nullptr, Vb, (int)MI, 768, 768);
  k_attn<<<BB * HH, 128, 0, stream>>>(Qb, Kb, Vb, updb);
  gemm(updb, wob, bo, Of, nullptr, (int)MT, 768, 768);

  gemm(textb, Wtb, nullptr, nullptr, WhTb, (int)MT, 768, 768);
  gemm(imgb,  Wib, nullptr, nullptr, WhIb, (int)MI, 768, 768);
  k_srcdst<<<(int)((MT * HH + 255) / 256), 256, 0, stream>>>(WhTb, at, srcT, dstT, (int)MT);
  k_srcdst<<<(int)((MI * HH + 255) / 256), 256, 0, stream>>>(WhIb, ai, srcI, dstI, (int)MI);
  k_gat<<<BB * HH, 128, 0, stream>>>(WhTb, srcT, dstT, tadj, TT * TT, tgfa, TT, 64, 2);
  k_gat<<<BB * HH, 128, 0, stream>>>(WhIb, srcI, dstI, iadj, 0, igfa, PPN, PPAD, 7);

  gemm(ctb, Wpb, bp, fg, nullptr, 32, 768, 512);
  k_r1<<<BB, 256, 0, stream>>>(Of, img, tgfa, igfa, ct, ci, ls, ftok, sc3);
  k_r2<<<BB, 256, 0, stream>>>(sc3, f1w, f1b, f2w, f2b, ftok, tgfa, fg, faggb);
  gemm(faggb, c1wb, c1b, h1, nullptr, 32, 384, 768);
  k_r3<<<1, 64, 0, stream>>>(h1, c2w, c2b, (float*)d_out);
}
